// AttentionMechanism_74483322847559
// MI455X (gfx1250) — matrix operand order repaired and checked by reading, not compiled, not run on hardware
//
#include <hip/hip_runtime.h>
#include <math.h>

// Problem constants (from reference)
#define B_      64
#define KLEN    4096
#define KDIM    256
#define QDIM    256
#define VDIM    256
#define ADIM    128
#define CONV_CH 10
#define CONV_K  201
#define KPAD    224          // conv K padded up to multiple of 32 (rows >=201 of F are zero)
#define ROWS    64           // t-rows per block in the e-kernel (4 waves x 16)
#define APL     (ROWS + 200 + 24)   // staged aw_prev window + zero tail = 288

typedef _Float16 v16h __attribute__((ext_vector_type(16)));
typedef _Float16 h8   __attribute__((ext_vector_type(8)));
typedef float    v8f  __attribute__((ext_vector_type(8)));

// ---------------- prep kernels (tiny, run once per call) ----------------

// WkT[a][k] = f16(Wk[k][a])   (128 x 256)
__global__ void prep_wk(const float* __restrict__ Wk, _Float16* __restrict__ WkT) {
    int idx = blockIdx.x * 256 + threadIdx.x;       // 128*256 = 32768 exact
    int a = idx >> 8, k = idx & 255;
    WkT[a * KDIM + k] = (_Float16)Wk[k * ADIM + a];
}

// Ft[a][j] = f16( sum_c conv_w[j][c] * Wc[c][a] ),  j>=201 -> 0   (128 x 224)
__global__ void prep_ft(const float* __restrict__ conv_w, const float* __restrict__ Wc,
                        _Float16* __restrict__ Ft) {
    int idx = blockIdx.x * 256 + threadIdx.x;       // 112*256 = 28672 = 128*224 exact
    int a = idx / KPAD, j = idx % KPAD;
    float s = 0.f;
    if (j < CONV_K) {
        #pragma unroll
        for (int c = 0; c < CONV_CH; ++c)
            s += conv_w[j * CONV_CH + c] * Wc[c * ADIM + a];
    }
    Ft[a * KPAD + j] = (_Float16)s;
}

// qb[b][a] = query[b] @ Wq[:,a] + bk[a]   (bias folded in)
__global__ void prep_q(const float* __restrict__ query, const float* __restrict__ Wq,
                       const float* __restrict__ bk, float* __restrict__ qb) {
    int idx = blockIdx.x * 256 + threadIdx.x;       // 32*256 = 8192 = 64*128 exact
    int b = idx >> 7, a = idx & 127;
    float s = bk[a];
    for (int k = 0; k < QDIM; ++k)
        s += query[b * QDIM + k] * Wq[k * ADIM + a];
    qb[idx] = s;
}

// ---------------- main fused e-kernel (WMMA) ----------------
// e[b,t] = sum_a Vw[a] * tanh( (key@Wk)[t,a] + qb[b,a] + (conv window @ F)[t,a] )
__global__ void __launch_bounds__(128)
attn_e_kernel(const float* __restrict__ key, const float* __restrict__ aw_prev,
              const _Float16* __restrict__ WkT, const _Float16* __restrict__ Ft,
              const float* __restrict__ qb, const float* __restrict__ Vw,
              float* __restrict__ e_out) {
    __shared__ float ap[APL];

    const int b  = blockIdx.x / (KLEN / ROWS);
    const int t0 = (blockIdx.x % (KLEN / ROWS)) * ROWS;
    const int tid = threadIdx.x;

    // Stage aw_prev[t0-100 .. t0+ROWS+99] (zero outside sequence; zero tail for K-padding)
    for (int i = tid; i < APL; i += 128) {
        int gt = t0 - 100 + i;
        float v = 0.f;
        if (i < ROWS + 200 && gt >= 0 && gt < KLEN) v = aw_prev[b * KLEN + gt];
        ap[i] = v;
    }
    __syncthreads();

    const int w    = tid >> 5;      // wave id, rows t0+16w .. t0+16w+15
    const int lane = tid & 31;
    const int half = lane >> 4;
    const int l15  = lane & 15;

    const v8f vzero = {0.f, 0.f, 0.f, 0.f, 0.f, 0.f, 0.f, 0.f};
    v8f acc[8];
    #pragma unroll
    for (int n = 0; n < 8; ++n) acc[n] = vzero;

    // ---- conv GEMM: A = sliding window of aw_prev (16 x 224), B = Ft tiles ----
    for (int kb = 0; kb < KPAD; kb += 32) {
        v16h af;
        const int base = 16 * w + l15 + kb + half * 8;   // LDS index of K element 0
        #pragma unroll
        for (int e = 0; e < 8; ++e) {
            af[e]     = (_Float16)ap[base + e];
            af[e + 8] = (_Float16)ap[base + 16 + e];
        }
        #pragma unroll
        for (int n = 0; n < 8; ++n) {
            const _Float16* frow = Ft + (16 * n + l15) * KPAD + kb + half * 8;
            h8 b0 = *(const h8*)(frow);
            h8 b1 = *(const h8*)(frow + 16);
            v16h bf = __builtin_shufflevector(b0, b1, 0,1,2,3,4,5,6,7,8,9,10,11,12,13,14,15);
            acc[n] = __builtin_amdgcn_wmma_f32_16x16x32_f16(
                false, af, false, bf, (short)0, acc[n], false, false);
        }
    }

    // ---- key GEMM: A = key rows (16 x 256 f32 -> f16), B = WkT tiles ----
    const float* krow = key + ((size_t)b * KLEN + (t0 + 16 * w + l15)) * KDIM;
    for (int kb = 0; kb < KDIM; kb += 32) {
        const int k0 = kb + half * 8;
        float4 x0 = *(const float4*)(krow + k0);
        float4 x1 = *(const float4*)(krow + k0 + 4);
        float4 x2 = *(const float4*)(krow + k0 + 16);
        float4 x3 = *(const float4*)(krow + k0 + 20);
        v16h af;
        af[0]  = (_Float16)x0.x; af[1]  = (_Float16)x0.y; af[2]  = (_Float16)x0.z; af[3]  = (_Float16)x0.w;
        af[4]  = (_Float16)x1.x; af[5]  = (_Float16)x1.y; af[6]  = (_Float16)x1.z; af[7]  = (_Float16)x1.w;
        af[8]  = (_Float16)x2.x; af[9]  = (_Float16)x2.y; af[10] = (_Float16)x2.z; af[11] = (_Float16)x2.w;
        af[12] = (_Float16)x3.x; af[13] = (_Float16)x3.y; af[14] = (_Float16)x3.z; af[15] = (_Float16)x3.w;
        #pragma unroll
        for (int n = 0; n < 8; ++n) {
            const _Float16* wrow = WkT + (16 * n + l15) * KDIM + kb + half * 8;
            h8 b0 = *(const h8*)(wrow);
            h8 b1 = *(const h8*)(wrow + 16);
            v16h bf = __builtin_shufflevector(b0, b1, 0,1,2,3,4,5,6,7,8,9,10,11,12,13,14,15);
            acc[n] = __builtin_amdgcn_wmma_f32_16x16x32_f16(
                false, af, false, bf, (short)0, acc[n], false, false);
        }
    }

    // ---- epilogue: +q, tanh, *Vw, reduce over adim ----
    // C/D layout: lane holds rows M = half*8 + r (r=0..7), column a = 16n + l15
    float es[8];
    #pragma unroll
    for (int r = 0; r < 8; ++r) es[r] = 0.f;
    #pragma unroll
    for (int n = 0; n < 8; ++n) {
        const int a = 16 * n + l15;
        const float qa = qb[b * ADIM + a];
        const float vw = Vw[a];
        #pragma unroll
        for (int r = 0; r < 8; ++r)
            es[r] += vw * tanhf(acc[n][r] + qa);
    }
    // sum across the 16 lanes (N columns) that share each row
    #pragma unroll
    for (int r = 0; r < 8; ++r) {
        float s = es[r];
        s += __shfl_xor(s, 1, 16);
        s += __shfl_xor(s, 2, 16);
        s += __shfl_xor(s, 4, 16);
        s += __shfl_xor(s, 8, 16);
        es[r] = s;
    }
    if (l15 == 0) {
        const int rowbase = t0 + 16 * w + half * 8;
        #pragma unroll
        for (int r = 0; r < 8; ++r)
            e_out[b * KLEN + rowbase + r] = es[r];
    }
}

// ---------------- softmax over klen, in place (block per b) ----------------
__global__ void softmax_kernel(float* __restrict__ aw) {
    const int b = blockIdx.x;
    float* p = aw + (size_t)b * KLEN;
    __shared__ float red[256];
    const int tid = threadIdx.x;

    float mx = -3.4e38f;
    for (int i = tid; i < KLEN; i += 256) mx = fmaxf(mx, p[i]);
    red[tid] = mx; __syncthreads();
    for (int s = 128; s > 0; s >>= 1) {
        if (tid < s) red[tid] = fmaxf(red[tid], red[tid + s]);
        __syncthreads();
    }
    mx = red[0]; __syncthreads();

    float sum = 0.f;
    for (int i = tid; i < KLEN; i += 256) {
        float v = expf(p[i] - mx);
        p[i] = v;
        sum += v;
    }
    red[tid] = sum; __syncthreads();
    for (int s = 128; s > 0; s >>= 1) {
        if (tid < s) red[tid] += red[tid + s];
        __syncthreads();
    }
    const float inv = 1.f / red[0];
    for (int i = tid; i < KLEN; i += 256) p[i] *= inv;
}

// ---------------- cv = aw @ value, deterministic 2-stage ----------------
__global__ void cv_partial(const float* __restrict__ aw, const float* __restrict__ value,
                           float* __restrict__ part) {
    const int b  = blockIdx.x >> 3;
    const int ch = blockIdx.x & 7;          // 8 chunks of 512 t-values
    __shared__ float awn[512];
    const int tid = threadIdx.x;
    const int tbase = ch * 512;
    awn[tid]       = aw[(size_t)b * KLEN + tbase + tid];
    awn[tid + 256] = aw[(size_t)b * KLEN + tbase + 256 + tid];
    __syncthreads();
    const float* vp = value + ((size_t)b * KLEN + tbase) * VDIM + tid;
    float s = 0.f;
    #pragma unroll 4
    for (int t = 0; t < 512; ++t)
        s += awn[t] * vp[(size_t)t * VDIM];
    part[((size_t)b * 8 + ch) * VDIM + tid] = s;
}

__global__ void cv_reduce(const float* __restrict__ part, float* __restrict__ cv) {
    const int idx = blockIdx.x * 256 + threadIdx.x;   // 64*256 = B*VDIM
    const int b = idx >> 8, v = idx & 255;
    float s = 0.f;
    #pragma unroll
    for (int c = 0; c < 8; ++c)
        s += part[((size_t)b * 8 + c) * VDIM + v];
    cv[idx] = s;
}

// ---------------- launcher ----------------
extern "C" void kernel_launch(void* const* d_in, const int* in_sizes, int n_in,
                              void* d_out, int out_size, void* d_ws, size_t ws_size,
                              hipStream_t stream) {
    const float* key    = (const float*)d_in[0];
    const float* value  = (const float*)d_in[1];
    const float* query  = (const float*)d_in[2];
    const float* awp    = (const float*)d_in[3];
    const float* Wk     = (const float*)d_in[4];
    const float* bk     = (const float*)d_in[5];
    const float* Wq     = (const float*)d_in[6];
    const float* conv_w = (const float*)d_in[7];
    const float* Wc     = (const float*)d_in[8];
    const float* Vw     = (const float*)d_in[9];

    float* out = (float*)d_out;
    float* cv  = out;                       // [B,1,VDIM]
    float* aw  = out + B_ * VDIM;           // [B,1,1,KLEN]  (holds e, then aw in place)

    char* ws = (char*)d_ws;
    _Float16* WkT  = (_Float16*)(ws + 0);        //  64 KB
    _Float16* Ft   = (_Float16*)(ws + 65536);    //  56 KB
    float*    qb   = (float*)(ws + 122880);      //  32 KB
    float*    part = (float*)(ws + 155648);      // 512 KB

    prep_wk<<<128, 256, 0, stream>>>(Wk, WkT);
    prep_ft<<<112, 256, 0, stream>>>(conv_w, Wc, Ft);
    prep_q <<<32,  256, 0, stream>>>(query, Wq, bk, qb);

    attn_e_kernel<<<B_ * (KLEN / ROWS), 128, 0, stream>>>(key, awp, WkT, Ft, qb, Vw, aw);
    softmax_kernel<<<B_, 256, 0, stream>>>(aw);
    cv_partial<<<B_ * 8, 256, 0, stream>>>(aw, value, part);
    cv_reduce<<<64, 256, 0, stream>>>(part, cv);
}
